// SASAConv2d_7198365188210
// MI455X (gfx1250) — matrix operand order repaired and checked by reading, not compiled, not run on hardware
//
#include <hip/hip_runtime.h>

// ---------------------------------------------------------------------------
// SASA local attention for MI455X (gfx1250, wave32, WMMA + TDM).
// Stage 0: f32 -> bf16 conversion of x (pixel-major) and Wq/Wk/Wv.
// Stage 1: q/k/v = W @ x. W staged to LDS via TENSOR_LOAD_TO_LDS (TDM);
//          each wave computes 16 pixels x 128 channels = 32 WMMAs.
// Stage 2: per-pixel 7x7 windowed attention, online softmax, VALU.
// ---------------------------------------------------------------------------

typedef __attribute__((ext_vector_type(16))) __bf16        v16bf;
typedef __attribute__((ext_vector_type(8)))  float         v8f;
typedef __attribute__((ext_vector_type(8)))  unsigned int  v8u;
typedef __attribute__((ext_vector_type(4)))  unsigned int  u32x4;
typedef __attribute__((ext_vector_type(8)))  int           i32x8;
typedef __attribute__((ext_vector_type(4)))  int           i32x4;

constexpr int N_IMG = 8;
constexpr int C     = 128;          // Cin == Cout
constexpr int H     = 56, W = 56;
constexpr int P     = H * W;        // 3136 pixels per image (196 tiles of 16)
constexpr int KW    = 7;            // window
constexpr int HEADS = 8, DH = 16;   // 8 heads x 16 dims

#if defined(__has_builtin)
#if __has_builtin(__builtin_amdgcn_tensor_load_to_lds) && \
    __has_builtin(__builtin_amdgcn_s_wait_tensorcnt)
#define SASA_HAVE_TDM 1
#endif
#endif

__device__ __forceinline__ unsigned short f2bf(float f) {
  unsigned int u = __float_as_uint(f);
  u += 0x7FFFu + ((u >> 16) & 1u);  // round-to-nearest-even
  return (unsigned short)(u >> 16);
}

// ---- Stage 0a: x [N][C][P] f32 -> xbf [N][P][C] bf16 (writes coalesced) ----
__global__ __launch_bounds__(256) void convert_x(const float* __restrict__ x,
                                                 unsigned short* __restrict__ xbf) {
  int idx = blockIdx.x * 256 + threadIdx.x;       // over [N][P][C/2] pairs
  if (idx >= N_IMG * P * (C / 2)) return;
  int pair = idx & 63;                            // C/2 = 64
  int t    = idx >> 6;                            // n*P + p
  int p = t % P, n = t / P;
  int c0 = pair * 2;
  float a = x[((size_t)n * C + c0    ) * P + p];
  float b = x[((size_t)n * C + c0 + 1) * P + p];
  ((unsigned int*)xbf)[idx] = (unsigned int)f2bf(a) | ((unsigned int)f2bf(b) << 16);
}

// ---- Stage 0b: Wq/Wk/Wv -> wbf [3][C][C] bf16 ----
__global__ __launch_bounds__(256) void convert_w(const float* __restrict__ wq,
                                                 const float* __restrict__ wk,
                                                 const float* __restrict__ wv,
                                                 unsigned short* __restrict__ wbf) {
  int idx = blockIdx.x * 256 + threadIdx.x;
  if (idx >= 3 * C * C) return;
  int s = idx / (C * C);
  int e = idx % (C * C);
  const float* src = (s == 0) ? wq : (s == 1) ? wk : wv;
  wbf[idx] = f2bf(src[e]);
}

// ---- Stage 1: qkv[s][n][p][c] = sum_k W[s][c][k] * x[n][p][k]  (WMMA) ----
// Block: stages W[s] (128x128 bf16 = 32 KB) into LDS via TDM, then each of
// 8 waves computes one 16-pixel tile across all 128 output channels:
// 8 row-tiles x 4 K-steps = 32 v_wmma per wave. B fragments (4 K-steps,
// 32 VGPRs) are loaded once and reused across all 8 row tiles.
__global__ __launch_bounds__(256) void qkv_gemm_wmma(
    const unsigned short* __restrict__ xbf,   // [N][P][C]  bf16
    const unsigned short* __restrict__ wbf,   // [3][C][C]  bf16 (row: out-chan)
    float* __restrict__ qkv) {                // [3][N][P][C] f32
  __shared__ __align__(16) unsigned short sW[C * C];   // 32 KB

  int s = blockIdx.y % 3;
  int n = blockIdx.y / 3;

#if SASA_HAVE_TDM
  if (threadIdx.x < 32) {  // wave 0 issues one DMA for the whole 128x128 tile
    unsigned long long ga =
        (unsigned long long)(uintptr_t)(wbf + (size_t)s * C * C);
    unsigned int lds_off = (unsigned int)(uintptr_t)&sW[0];  // addr[31:0] = LDS offset
    // D# group 0: count=1 (valid), lds_addr, global_addr[56:0], type=2.
    u32x4 g0;
    g0[0] = 1u;
    g0[1] = lds_off;
    g0[2] = (unsigned int)ga;
    g0[3] = ((unsigned int)(ga >> 32) & 0x01FFFFFFu) | (2u << 30);
    // D# group 1: data_size=2B; tensor 128x128; tile 128x128; dim0 stride 128.
    i32x8 g1;
    g1[0] = (1 << 16);           // data_size=1 (2 bytes); no mask/flags/pad
    g1[1] = (int)(128u << 16);   // tensor_dim0[15:0] at bits [63:48]
    g1[2] = (int)(128u << 16);   // tensor_dim1[15:0] at bits [95:80]
    g1[3] = (int)(128u << 16);   // tile_dim0 at bits [127:112]
    g1[4] = 128;                 // tile_dim1 at bits [143:128]
    g1[5] = 128;                 // tensor_dim0_stride[31:0] at bits [191:160]
    g1[6] = 0;
    g1[7] = 0;
    i32x4 z4 = {};
#if __clang_major__ >= 23
    i32x8 z8 = {};
    __builtin_amdgcn_tensor_load_to_lds(g0, g1, z4, z4, z8, 0);
#else
    __builtin_amdgcn_tensor_load_to_lds(g0, g1, z4, z4, 0);
#endif
    __builtin_amdgcn_s_wait_tensorcnt(0);
  }
  __syncthreads();
#else
  {  // fallback: cooperative 16B copies
    const uint4* src = (const uint4*)(wbf + (size_t)s * C * C);
    for (int i = threadIdx.x; i < C * C / 8; i += 256) ((uint4*)sW)[i] = src[i];
  }
  __syncthreads();
#endif

  int lane = threadIdx.x & 31;
  int wave = threadIdx.x >> 5;
  int tile = blockIdx.x * 8 + wave;           // pixel tile id, 0..195 valid
  int l15 = lane & 15;
  int g   = lane >> 4;

  if (tile < P / 16) {
    int col0 = tile * 16;                     // pixel base
    const unsigned short* X = xbf + (size_t)n * P * C;

    // B fragments for all 4 K-steps: 32x16 bf16, col N = l15;
    // half-lane g holds K {16g..16g+15} as contiguous pairs -> 2x b128 each.
    v8u bu[4];
#pragma unroll
    for (int ks = 0; ks < 4; ++ks) {
      const unsigned short* Brow =
          X + (size_t)(col0 + l15) * C + ks * 32 + 16 * g;
#pragma unroll
      for (int v = 0; v < 8; ++v)
        bu[ks][v] = *(const unsigned int*)(Brow + 2 * v);
    }

#pragma unroll
    for (int rt = 0; rt < 8; ++rt) {
      int row0 = rt * 16;                     // output-channel base
      v8f acc = {};
#pragma unroll
      for (int ks = 0; ks < 4; ++ks) {
        // A fragment from LDS: 16x32 bf16, row M = l15;
        // half-lane g holds K {16g..16g+15} -> 2x ds_load_b128.
        const unsigned short* Arow = sW + (row0 + l15) * C + ks * 32;
        uint4 a0 = *(const uint4*)(Arow + 16 * g);
        uint4 a1 = *(const uint4*)(Arow + 16 * g + 8);
        v8u au;
        au[0] = a0.x; au[1] = a0.y; au[2] = a0.z; au[3] = a0.w;
        au[4] = a1.x; au[5] = a1.y; au[6] = a1.z; au[7] = a1.w;
        acc = __builtin_amdgcn_wmma_f32_16x16x32_bf16(
            false, __builtin_bit_cast(v16bf, au),
            false, __builtin_bit_cast(v16bf, bu[ks]),
            (short)0, acc, false, false);
      }
      // C/D layout: VGPR r -> row (row0 + r + 8g), lane15 -> pixel column.
      float* Cp = qkv + (((size_t)s * N_IMG + n) * P + (col0 + l15)) * C +
                  row0 + 8 * g;
      float4 lo, hi;
      lo.x = acc[0]; lo.y = acc[1]; lo.z = acc[2]; lo.w = acc[3];
      hi.x = acc[4]; hi.y = acc[5]; hi.z = acc[6]; hi.w = acc[7];
      *(float4*)(Cp)     = lo;
      *(float4*)(Cp + 4) = hi;
    }
  }
}

// ---- Stage 2: 7x7 windowed attention, online softmax, one thread/(n,m,p) ----
__global__ __launch_bounds__(256) void sasa_attention(
    const float* __restrict__ qkv,            // [3][N][P][C]
    const float* __restrict__ h_emb,          // [64][7]
    const float* __restrict__ w_emb,          // [64][7]
    float* __restrict__ out) {                // [N][C][P]
  const size_t NPC = (size_t)N_IMG * P * C;
  int nm = blockIdx.y;                        // n*HEADS + m
  int n = nm >> 3, m = nm & 7;

  __shared__ float s_rel[DH * KW];            // rel bias for this head
  if (threadIdx.x < DH * KW) {
    int d = threadIdx.x / KW, r = threadIdx.x % KW;
    int c = m * DH + d;
    s_rel[threadIdx.x] = (c < C / 2) ? h_emb[c * KW + r]
                                     : w_emb[(c - C / 2) * KW + r];
  }
  __syncthreads();
  const bool use_i = (m < 4);                 // first half varies with kernel row

  int p = blockIdx.x * 256 + threadIdx.x;
  if (p >= P) return;
  int h = p / W, w = p % W;

  const float* qp = qkv + ((size_t)n * P + p) * C + m * DH;
  float q[DH];
#pragma unroll
  for (int d = 0; d < DH; ++d) q[d] = qp[d];

  float mx = -3.0e38f, denom = 0.f, acc[DH];
#pragma unroll
  for (int d = 0; d < DH; ++d) acc[d] = 0.f;

  const float* kbase = qkv + NPC;
  const float* vbase = qkv + 2 * NPC;

  for (int i = 0; i < KW; ++i) {
    int hh = h + i - (KW - 1) / 2;
    for (int j = 0; j < KW; ++j) {
      int ww = w + j - (KW - 1) / 2;
      bool inb = (hh >= 0) & (hh < H) & (ww >= 0) & (ww < W);
      int pp = inb ? (hh * W + ww) : 0;
      const float* kp = kbase + ((size_t)n * P + pp) * C + m * DH;
      const float* vp = vbase + ((size_t)n * P + pp) * C + m * DH;
      int ri = use_i ? i : j;
      float sc = 0.f, vv[DH];
#pragma unroll
      for (int d = 0; d < DH; ++d) {
        // padded positions: k patch is zero but rel bias still added (reference)
        float kd = (inb ? kp[d] : 0.f) + s_rel[d * KW + ri];
        sc = fmaf(q[d], kd, sc);
        vv[d] = inb ? vp[d] : 0.f;            // v patch: zero, no bias
      }
      float nmx   = fmaxf(mx, sc);
      float scale = __expf(mx - nmx);
      float e     = __expf(sc - nmx);
      denom = denom * scale + e;              // OOB weight still in denominator
#pragma unroll
      for (int d = 0; d < DH; ++d) acc[d] = fmaf(acc[d], scale, e * vv[d]);
      mx = nmx;
    }
  }
  float inv = 1.0f / denom;
  float* op = out + ((size_t)n * C + m * DH) * P + p;   // coalesced across lanes
#pragma unroll
  for (int d = 0; d < DH; ++d) op[(size_t)d * P] = acc[d] * inv;
}

// ---------------------------------------------------------------------------
extern "C" void kernel_launch(void* const* d_in, const int* in_sizes, int n_in,
                              void* d_out, int out_size, void* d_ws, size_t ws_size,
                              hipStream_t stream) {
  const float* x    = (const float*)d_in[0];
  const float* Wq   = (const float*)d_in[1];
  const float* Wk   = (const float*)d_in[2];
  const float* Wv   = (const float*)d_in[3];
  const float* hemb = (const float*)d_in[4];
  const float* wemb = (const float*)d_in[5];
  float* out = (float*)d_out;

  // Workspace layout:
  //   xbf : [N][P][C]     bf16   6.4 MB
  //   wbf : [3][C][C]     bf16   0.1 MB
  //   qkv : [3][N][P][C]  f32   38.5 MB
  unsigned short* xbf = (unsigned short*)d_ws;
  unsigned short* wbf = xbf + (size_t)N_IMG * P * C;
  float* qkv = (float*)(wbf + 3 * C * C);

  convert_x<<<dim3((N_IMG * P * (C / 2) + 255) / 256), 256, 0, stream>>>(x, xbf);
  convert_w<<<dim3((3 * C * C + 255) / 256), 256, 0, stream>>>(Wq, Wk, Wv, wbf);

  // 196 pixel tiles, 8 waves/block -> 25 blocks (last partially active).
  qkv_gemm_wmma<<<dim3((P / 16 + 7) / 8, N_IMG * 3), 256, 0, stream>>>(
      xbf, wbf, qkv);

  sasa_attention<<<dim3((P + 255) / 256, N_IMG * HEADS), 256, 0, stream>>>(
      qkv, hemb, wemb, out);
}
